// MultiHeadSelfAttention_7842610283008
// MI455X (gfx1250) — matrix operand order repaired and checked by reading, not compiled, not run on hardware
//
#include <hip/hip_runtime.h>

// ---------------------------------------------------------------------------
// Multi-head self-attention forward for MI455X (gfx1250), wave32 + WMMA bf16.
// B=4, S=2048, E=1024, H=16, D=64.  M = B*S = 8192.
// Software-pipelined WMMA GEMMs: next K-step loads issue before current
// K-step's WMMAs so the matrix pipe overlaps with L2/HBM latency.
// ---------------------------------------------------------------------------

typedef __bf16 bf16;
typedef bf16  v16bf __attribute__((ext_vector_type(16)));
typedef bf16  v8bf  __attribute__((ext_vector_type(8)));
typedef float v8f   __attribute__((ext_vector_type(8)));
typedef float v4f   __attribute__((ext_vector_type(4)));

#define EMBED 1024
#define HEADS 16
#define HDIM  64
#define SEQ   2048
#define BATCH 4
#define MTOT  (BATCH * SEQ)      // 8192

// ---------------- fragment loaders (layouts per CDNA5 ISA 7.12.2) ----------

// A fragment 16x32 bf16: lane L holds row (L&15); K-chunks at k0+(L>>4)*8 and +16.
__device__ __forceinline__ v16bf load_a_frag(const bf16* __restrict__ A,
                                             int row0, int k0, int lda) {
  int lane = threadIdx.x & 31;
  const bf16* p = A + (size_t)(row0 + (lane & 15)) * lda + k0 + ((lane >> 4) << 3);
  union { v16bf v; v8bf h[2]; } u;
  u.h[0] = *(const v8bf*)(p);
  u.h[1] = *(const v8bf*)(p + 16);
  return u.v;
}

// Same, but source is f32 (softmax probabilities) converted to bf16 in-register.
__device__ __forceinline__ v16bf load_a_frag_f32(const float* __restrict__ A,
                                                 int row0, int k0, int lda) {
  int lane = threadIdx.x & 31;
  const float* p = A + (size_t)(row0 + (lane & 15)) * lda + k0 + ((lane >> 4) << 3);
  v8f f0 = *(const v8f*)(p);
  v8f f1 = *(const v8f*)(p + 16);
  union { v16bf v; bf16 e[16]; } u;
#pragma unroll
  for (int i = 0; i < 8; ++i) { u.e[i] = (bf16)f0[i]; u.e[8 + i] = (bf16)f1[i]; }
  return u.v;
}

// B fragment 32x16 bf16 from Bt stored [N,K] row-major: lane L holds column
// n0+(L&15); K-chunks at k0+(L>>4)*8 and +16 (same order as A). Two 16B loads.
__device__ __forceinline__ v16bf load_b_frag(const bf16* __restrict__ Bt,
                                             int n0, int k0, int ldb) {
  int lane = threadIdx.x & 31;
  const bf16* p = Bt + (size_t)(n0 + (lane & 15)) * ldb + k0 + ((lane >> 4) << 3);
  union { v16bf v; v8bf h[2]; } u;
  u.h[0] = *(const v8bf*)(p);
  u.h[1] = *(const v8bf*)(p + 16);
  return u.v;
}

#define WMMA_BF16(a, b, c) \
  __builtin_amdgcn_wmma_f32_16x16x32_bf16(false, (a), false, (b), (short)0, (c), false, false)

// ---------------- conversion / transpose kernels ---------------------------

__global__ void cvt_f32_bf16_kernel(const float* __restrict__ in,
                                    bf16* __restrict__ out, int n) {
  int i = (blockIdx.x * blockDim.x + threadIdx.x) * 4;
  if (i + 3 < n) {
    v4f f = *(const v4f*)(in + i);
#pragma unroll
    for (int j = 0; j < 4; ++j) out[i + j] = (bf16)f[j];
  }
}

// W [K=1024, N=1024] f32 -> Wt [N, K] bf16, LDS-tiled.
__global__ void transpose_cvt_kernel(const float* __restrict__ W,
                                     bf16* __restrict__ Wt) {
  __shared__ float tile[32][33];
  int c0 = blockIdx.x * 32, r0 = blockIdx.y * 32;
#pragma unroll
  for (int rr = threadIdx.y; rr < 32; rr += 8)
    tile[rr][threadIdx.x] = W[(size_t)(r0 + rr) * EMBED + c0 + threadIdx.x];
  __syncthreads();
#pragma unroll
  for (int rr = threadIdx.y; rr < 32; rr += 8)
    Wt[(size_t)(c0 + rr) * EMBED + r0 + threadIdx.x] = (bf16)tile[threadIdx.x][rr];
}

// ---------------- GEMM: [M,1024] bf16 @ Wt[1024,1024] + bias ---------------
// mode 0: out bf16 at [B,H,S,D]   (q, k)
// mode 1: out bf16 at [B,H,D,S]   (v, pre-transposed for the P@V GEMM)
// mode 3: out f32  at [M,E]       (final output projection)
__global__ void gemm_proj_kernel(const bf16* __restrict__ A,
                                 const bf16* __restrict__ Bt,
                                 const float* __restrict__ bias,
                                 bf16* __restrict__ outb,
                                 float* __restrict__ outf, int mode) {
  int wave = threadIdx.x >> 5;
  int m0 = blockIdx.x * 128 + wave * 16;
  int n0 = blockIdx.y * 64;
  v8f acc[4] = {};

  // software pipeline: stage K-step k0 while computing k0-32
  v16bf ac = load_a_frag(A, m0, 0, EMBED);
  v16bf bc[4];
#pragma unroll
  for (int t = 0; t < 4; ++t) bc[t] = load_b_frag(Bt, n0 + 16 * t, 0, EMBED);

#pragma unroll 4
  for (int k0 = 32; k0 < EMBED; k0 += 32) {
    __builtin_prefetch(A + (size_t)m0 * EMBED + k0 + 64, 0, 1);
    v16bf an = load_a_frag(A, m0, k0, EMBED);
    v16bf bn[4];
#pragma unroll
    for (int t = 0; t < 4; ++t) bn[t] = load_b_frag(Bt, n0 + 16 * t, k0, EMBED);
#pragma unroll
    for (int t = 0; t < 4; ++t) acc[t] = WMMA_BF16(ac, bc[t], acc[t]);
    ac = an;
#pragma unroll
    for (int t = 0; t < 4; ++t) bc[t] = bn[t];
  }
#pragma unroll
  for (int t = 0; t < 4; ++t) acc[t] = WMMA_BF16(ac, bc[t], acc[t]);

  // epilogue: C layout — VGPR r: lanes 0-15 -> M=r, lanes 16-31 -> M=r+8; N = lane&15
  int lane = threadIdx.x & 31;
  int nn = lane & 15;
  int mo = (lane >> 4) * 8;
#pragma unroll
  for (int t = 0; t < 4; ++t) {
    int n = n0 + t * 16 + nn;
    float bs = bias[n];
#pragma unroll
    for (int r = 0; r < 8; ++r) {
      int m = m0 + r + mo;
      float val = acc[t][r] + bs;
      if (mode == 3) {
        outf[(size_t)m * EMBED + n] = val;
      } else {
        int bb = m >> 11, s = m & (SEQ - 1);
        int h = n >> 6, d = n & (HDIM - 1);
        size_t idx = (mode == 0)
            ? ((size_t)(bb * HEADS + h) * SEQ + s) * HDIM + d
            : ((size_t)(bb * HEADS + h) * HDIM + d) * SEQ + s;
        outb[idx] = (bf16)val;
      }
    }
  }
}

// ---------------- scores: q @ k^T * (1/sqrt(D)) ----------------------------
// grid (S/128 colblocks, S/128 rowblocks, B*H), block 256 (8 waves, 16x128 each)
__global__ void scores_kernel(const bf16* __restrict__ Q,
                              const bf16* __restrict__ Km,
                              float* __restrict__ Sc) {
  int bh = blockIdx.z;
  const bf16* q = Q + (size_t)bh * SEQ * HDIM;
  const bf16* k = Km + (size_t)bh * SEQ * HDIM;
  float* sc = Sc + (size_t)bh * SEQ * SEQ;
  int wave = threadIdx.x >> 5;
  int sq0 = blockIdx.y * 128 + wave * 16;
  int sk0 = blockIdx.x * 128;
  v8f acc[8] = {};
#pragma unroll
  for (int d0 = 0; d0 < HDIM; d0 += 32) {
    // batch all loads of this K-step before the WMMA burst
    v16bf a = load_a_frag(q, sq0, d0, HDIM);
    v16bf bfr[8];
#pragma unroll
    for (int t = 0; t < 8; ++t) bfr[t] = load_b_frag(k, sk0 + 16 * t, d0, HDIM);
#pragma unroll
    for (int t = 0; t < 8; ++t) acc[t] = WMMA_BF16(a, bfr[t], acc[t]);
  }
  const float scale = 0.125f;  // 1/sqrt(64)
  int lane = threadIdx.x & 31;
  int nn = lane & 15;
  int mo = (lane >> 4) * 8;
#pragma unroll
  for (int t = 0; t < 8; ++t)
#pragma unroll
    for (int r = 0; r < 8; ++r)
      sc[(size_t)(sq0 + r + mo) * SEQ + sk0 + t * 16 + nn] = acc[t][r] * scale;
}

// ---------------- row softmax over 2048, in place --------------------------
__global__ void softmax_kernel(float* __restrict__ Sc) {
  __shared__ float red[256];
  float* p = Sc + ((size_t)blockIdx.y * SEQ + blockIdx.x) * SEQ;
  int tid = threadIdx.x;
  float vals[8];
  float mx = -3.402823466e+38f;
#pragma unroll
  for (int i = 0; i < 8; ++i) {
    vals[i] = p[tid + i * 256];
    mx = fmaxf(mx, vals[i]);
  }
  red[tid] = mx; __syncthreads();
  for (int s = 128; s > 0; s >>= 1) {
    if (tid < s) red[tid] = fmaxf(red[tid], red[tid + s]);
    __syncthreads();
  }
  mx = red[0]; __syncthreads();
  float sum = 0.f;
#pragma unroll
  for (int i = 0; i < 8; ++i) {
    vals[i] = __expf(vals[i] - mx);
    sum += vals[i];
  }
  red[tid] = sum; __syncthreads();
  for (int s = 128; s > 0; s >>= 1) {
    if (tid < s) red[tid] += red[tid + s];
    __syncthreads();
  }
  float inv = 1.f / red[0];
#pragma unroll
  for (int i = 0; i < 8; ++i) p[tid + i * 256] = vals[i] * inv;
}

// ---------------- attn_out = P @ V -----------------------------------------
// P f32 [S,S] per (b,h); Vt bf16 [B,H,D,S]; out bf16 merged-head [M,E].
// grid (S/128, B*H), block 256 (8 waves, 16 rows x 64 cols each)
__global__ void av_kernel(const float* __restrict__ P,
                          const bf16* __restrict__ Vt,
                          bf16* __restrict__ AO) {
  int bh = blockIdx.y;
  const float* pr = P + (size_t)bh * SEQ * SEQ;
  const bf16* vt = Vt + (size_t)bh * HDIM * SEQ;
  int wave = threadIdx.x >> 5;
  int sq0 = blockIdx.x * 128 + wave * 16;
  v8f acc[4] = {};

  v16bf ac = load_a_frag_f32(pr, sq0, 0, SEQ);
  v16bf bc[4];
#pragma unroll
  for (int t = 0; t < 4; ++t) bc[t] = load_b_frag(vt, t * 16, 0, SEQ);

#pragma unroll 4
  for (int k0 = 32; k0 < SEQ; k0 += 32) {
    __builtin_prefetch(pr + (size_t)sq0 * SEQ + k0 + 64, 0, 1);
    v16bf an = load_a_frag_f32(pr, sq0, k0, SEQ);
    v16bf bn[4];
#pragma unroll
    for (int t = 0; t < 4; ++t) bn[t] = load_b_frag(vt, t * 16, k0, SEQ);
#pragma unroll
    for (int t = 0; t < 4; ++t) acc[t] = WMMA_BF16(ac, bc[t], acc[t]);
    ac = an;
#pragma unroll
    for (int t = 0; t < 4; ++t) bc[t] = bn[t];
  }
#pragma unroll
  for (int t = 0; t < 4; ++t) acc[t] = WMMA_BF16(ac, bc[t], acc[t]);

  int lane = threadIdx.x & 31;
  int nn = lane & 15;
  int mo = (lane >> 4) * 8;
  int bb = bh >> 4, h = bh & 15;
#pragma unroll
  for (int t = 0; t < 4; ++t)
#pragma unroll
    for (int r = 0; r < 8; ++r) {
      int sq = sq0 + r + mo;
      int d = t * 16 + nn;
      AO[(size_t)(bb * SEQ + sq) * EMBED + h * HDIM + d] = (bf16)acc[t][r];
    }
}

// ---------------- launch ----------------------------------------------------

extern "C" void kernel_launch(void* const* d_in, const int* in_sizes, int n_in,
                              void* d_out, int out_size, void* d_ws, size_t ws_size,
                              hipStream_t stream) {
  const float* x  = (const float*)d_in[0];
  const float* wq = (const float*)d_in[1];
  const float* wk = (const float*)d_in[2];
  const float* wv = (const float*)d_in[3];
  const float* wo = (const float*)d_in[4];
  const float* bq = (const float*)d_in[5];
  const float* bk = (const float*)d_in[6];
  const float* bv = (const float*)d_in[7];
  const float* bo = (const float*)d_in[8];

  float* out   = (float*)d_out;                          // [4,2048,1024]
  float* attnW = out + (size_t)MTOT * EMBED;             // [4,16,2048,2048]

  char* ws = (char*)d_ws;
  const size_t SZ_XB = (size_t)MTOT * EMBED * 2;         // 16 MB
  const size_t SZ_W  = (size_t)EMBED * EMBED * 2;        // 2 MB
  const size_t SZ_QKV = (size_t)MTOT * EMBED * 2;        // 16 MB each
  bf16* XB  = (bf16*)(ws);
  bf16* WQT = (bf16*)(ws + SZ_XB);
  bf16* WKT = (bf16*)(ws + SZ_XB + SZ_W);
  bf16* WVT = (bf16*)(ws + SZ_XB + 2 * SZ_W);
  bf16* WOT = (bf16*)(ws + SZ_XB + 3 * SZ_W);
  bf16* Qb  = (bf16*)(ws + SZ_XB + 4 * SZ_W);
  bf16* Kb  = (bf16*)(ws + SZ_XB + 4 * SZ_W + SZ_QKV);
  bf16* VTb = (bf16*)(ws + SZ_XB + 4 * SZ_W + 2 * SZ_QKV);
  bf16* AOb = (bf16*)(ws + SZ_XB + 4 * SZ_W + 3 * SZ_QKV);

  // 1) precision staging
  cvt_f32_bf16_kernel<<<(MTOT * EMBED) / 1024, 256, 0, stream>>>(x, XB, MTOT * EMBED);
  transpose_cvt_kernel<<<dim3(32, 32), dim3(32, 8), 0, stream>>>(wq, WQT);
  transpose_cvt_kernel<<<dim3(32, 32), dim3(32, 8), 0, stream>>>(wk, WKT);
  transpose_cvt_kernel<<<dim3(32, 32), dim3(32, 8), 0, stream>>>(wv, WVT);
  transpose_cvt_kernel<<<dim3(32, 32), dim3(32, 8), 0, stream>>>(wo, WOT);

  // 2) QKV projections (WMMA bf16)
  gemm_proj_kernel<<<dim3(MTOT / 128, EMBED / 64), 256, 0, stream>>>(XB, WQT, bq, Qb,  nullptr, 0);
  gemm_proj_kernel<<<dim3(MTOT / 128, EMBED / 64), 256, 0, stream>>>(XB, WKT, bk, Kb,  nullptr, 0);
  gemm_proj_kernel<<<dim3(MTOT / 128, EMBED / 64), 256, 0, stream>>>(XB, WVT, bv, VTb, nullptr, 1);

  // 3) scores -> d_out attn region
  scores_kernel<<<dim3(SEQ / 128, SEQ / 128, BATCH * HEADS), 256, 0, stream>>>(Qb, Kb, attnW);

  // 4) softmax in place (final attn_weights output)
  softmax_kernel<<<dim3(SEQ, BATCH * HEADS), 256, 0, stream>>>(attnW);

  // 5) attn_out = P @ V (bf16 WMMA, P converted on the fly)
  av_kernel<<<dim3(SEQ / 128, BATCH * HEADS), 256, 0, stream>>>(attnW, VTb, AOb);

  // 6) output projection -> d_out
  gemm_proj_kernel<<<dim3(MTOT / 128, EMBED / 64), 256, 0, stream>>>(AOb, WOT, bo, nullptr, out, 3);
}
